// CausalSelfAttention_20289425507042
// MI455X (gfx1250) — matrix operand order repaired and checked by reading, not compiled, not run on hardware
//
#include <hip/hip_runtime.h>

// ---- problem constants ----
#define Bk    4
#define Tk    2048
#define Ck    2048
#define Hk    16
#define HKVk  4
#define Dk    128
#define Gk    4
#define BT    (Bk * Tk)        // 8192
#define KVC   (2 * HKVk * Dk)  // 1024

typedef unsigned int   u32;
typedef unsigned short u16;
typedef __attribute__((ext_vector_type(16))) __bf16 v16bf;
typedef __attribute__((ext_vector_type(8)))  float  v8f;
typedef __attribute__((ext_vector_type(4)))  u32    u32x4;

union Frag16 { u32x4 q[2]; v16bf v; };

__device__ __forceinline__ u16 f2bf(float f) {
  union { float f; u32 u; } a; a.f = f;
  u32 r = a.u + 0x7FFFu + ((a.u >> 16) & 1u);   // round-to-nearest-even
  return (u16)(r >> 16);
}

__device__ __forceinline__ v8f v8f_zero() {
  v8f v = {0.f, 0.f, 0.f, 0.f, 0.f, 0.f, 0.f, 0.f};
  return v;
}

// A fragment, 16x32 bf16 (ISA layout: lane<16 -> K {0..7,16..23}, lane>=16 -> +8)
__device__ __forceinline__ v16bf load_afrag(const u16* __restrict__ base, int ldk,
                                            int row, int kb, int lane) {
  const int aoff = (lane & 16) ? 8 : 0;
  const u16* p = base + (size_t)row * ldk + kb + aoff;
  Frag16 f;
  f.q[0] = *(const u32x4*)(p);
  f.q[1] = *(const u32x4*)(p + 16);
  return f.v;
}

// B fragment, 32x16 bf16 from K-contiguous storage (lane<16 -> K {0..7,16..23}, lane>=16 -> +8)
__device__ __forceinline__ v16bf load_bfrag(const u16* __restrict__ base, int ldk,
                                            int n, int kb, int lane) {
  const int koff = (lane & 16) ? 8 : 0;
  const u16* p = base + (size_t)n * ldk + kb + koff;
  Frag16 f;
  f.q[0] = *(const u32x4*)(p);
  f.q[1] = *(const u32x4*)(p + 16);
  return f.v;
}

// ---- elementwise convert / transpose kernels ----
__global__ void cvt_f32_bf16(const float* __restrict__ in, u16* __restrict__ out, int n) {
  int i = blockIdx.x * blockDim.x + threadIdx.x;
  if (i < n) out[i] = f2bf(in[i]);
}

// W [K,N] fp32 -> Wt [N,K] bf16
__global__ void transpose_cvt(const float* __restrict__ W, u16* __restrict__ Wt, int K, int N) {
  int i = blockIdx.x * blockDim.x + threadIdx.x;
  if (i >= N * K) return;
  int k = i % K, n = i / K;
  Wt[i] = f2bf(W[(size_t)k * N + n]);
}

// KVb [B*T, KVC] bf16 -> Vt [B*HKV*D, T] bf16 (V part only)
__global__ void transpose_v_kernel(const u16* __restrict__ KVb, u16* __restrict__ Vt, int n) {
  int i = blockIdx.x * blockDim.x + threadIdx.x;
  if (i >= n) return;
  int s = i % Tk;  int r = i / Tk;
  int d = r % Dk;  r /= Dk;
  int hk = r % HKVk;
  int b  = r / HKVk;
  Vt[i] = KVb[((size_t)(b * Tk + s)) * KVC + HKVk * Dk + hk * Dk + d];
}

// ---- WMMA GEMM: C[M,N] = A[M,K](bf16 row-major) * Bt[N,K]^T (bf16, K-contiguous) ----
// Each wave computes a 16x64 strip: 1 A-fragment feeds 4 WMMAs per K-step.
// MODE 0: store bf16, scaled.  MODE 1: store fp32 + bias.
template <int MODE>
__global__ __launch_bounds__(256) void gemm_wmma_kernel(
    const u16* __restrict__ A, const u16* __restrict__ Bt,
    u16* __restrict__ outB, float* __restrict__ outF, const float* __restrict__ bias,
    int M, int N, int K, float scale) {
  const int lane = threadIdx.x & 31;
  const int wave = threadIdx.x >> 5;
  const int nt = blockIdx.x * 64;
  const int mt = blockIdx.y * 128 + wave * 16;
  const int am = mt + (lane & 15);
  const int bn = nt + (lane & 15);
  v8f acc[4];
#pragma unroll
  for (int j = 0; j < 4; ++j) acc[j] = v8f_zero();

#pragma unroll 2
  for (int kb = 0; kb < K; kb += 32) {
    v16bf af = load_afrag(A, K, am, kb, lane);
    // speculative prefetch of the K-tile 128 elements (4 K-steps) ahead (global_prefetch_b8)
    if (kb + 128 < K) {
      __builtin_prefetch(A + (size_t)am * K + kb + 128, 0, 1);
      __builtin_prefetch(Bt + (size_t)bn * K + kb + 128, 0, 1);
    }
#pragma unroll
    for (int j = 0; j < 4; ++j) {
      v16bf bf = load_bfrag(Bt, K, bn + j * 16, kb, lane);
      acc[j] = __builtin_amdgcn_wmma_f32_16x16x32_bf16(false, af, false, bf, (short)0,
                                                       acc[j], false, false);
    }
  }
  const int roff = (lane & 16) ? 8 : 0;
#pragma unroll
  for (int j = 0; j < 4; ++j) {
    const int col = nt + j * 16 + (lane & 15);
#pragma unroll
    for (int i = 0; i < 8; ++i) {
      int row = mt + roff + i;
      float v = acc[j][i] * scale;
      if (MODE == 0) outB[(size_t)row * N + col] = f2bf(v);
      else           outF[(size_t)row * N + col] = v + bias[col];
    }
  }
}

// ---- Flash attention: one 16-row query tile per wave, 32-key tiles ----
__global__ __launch_bounds__(256) void attn_kernel(
    const u16* __restrict__ Qb,   // [B*T, C] bf16, pre-scaled by 1/sqrt(D)
    const u16* __restrict__ KVb,  // [B*T, KVC] bf16 (K cols 0..511, V cols 512..1023)
    const u16* __restrict__ Vt,   // [B*HKV*D, T] bf16
    u16* __restrict__ Yb) {       // [B*T, C] bf16
  __shared__ __align__(16) u16 psh[8][16 * 48];  // per-wave P tile, stride 48 halves
  const int lane = threadIdx.x & 31;
  const int wave = threadIdx.x >> 5;
  const int b = blockIdx.z;
  const int h = blockIdx.y;
  const int hk = h >> 2;  // G = 4
  const int qt0 = (blockIdx.x * 8 + wave) * 16;

  const u16* Qhead  = Qb  + (size_t)b * Tk * Ck  + (size_t)h  * Dk;
  const u16* Khead  = KVb + (size_t)b * Tk * KVC + (size_t)hk * Dk;
  const u16* Vthead = Vt  + ((size_t)(b * HKVk + hk)) * Dk * Tk;

  const int l15    = lane & 15;
  const int aoff   = (lane & 16) ? 8 : 0;
  const int roff   = (lane & 16) ? 8 : 0;
  const int koff16 = (lane & 16) ? 16 : 0;

  // Q fragments (16 x 128 split into 4 k-chunks of 32)
  v16bf qf[4];
#pragma unroll
  for (int c = 0; c < 4; ++c) {
    const u16* p = Qhead + (size_t)(qt0 + l15) * Ck + 32 * c + aoff;
    Frag16 f;
    f.q[0] = *(const u32x4*)(p);
    f.q[1] = *(const u32x4*)(p + 16);
    qf[c] = f.v;
  }

  float mrun[8], lrun[8], alpha[8];
  v8f o[8];
#pragma unroll
  for (int i = 0; i < 8; ++i) { mrun[i] = -3.0e38f; lrun[i] = 0.f; }
#pragma unroll
  for (int dc = 0; dc < 8; ++dc) o[dc] = v8f_zero();

  u16* pt = &psh[wave][0];

  for (int s0 = 0; s0 <= qt0 + 15; s0 += 32) {
    // prefetch next key tile's K/V rows while this tile computes
    if (s0 + 32 <= qt0 + 15) {
      __builtin_prefetch(Khead + (size_t)(s0 + 32 + l15) * KVC, 0, 1);
      __builtin_prefetch(Vthead + (size_t)l15 * Tk + s0 + 32, 0, 1);
    }
    // ---- S = Q K^T for 32 keys (two 16x16 column tiles) ----
    v8f sa = v8f_zero(), sb = v8f_zero();
#pragma unroll
    for (int c = 0; c < 4; ++c) {
      v16bf k0 = load_bfrag(Khead, KVC, s0 + l15,      32 * c, lane);
      v16bf k1 = load_bfrag(Khead, KVC, s0 + 16 + l15, 32 * c, lane);
      sa = __builtin_amdgcn_wmma_f32_16x16x32_bf16(false, qf[c], false, k0, (short)0, sa,
                                                   false, false);
      sb = __builtin_amdgcn_wmma_f32_16x16x32_bf16(false, qf[c], false, k1, (short)0, sb,
                                                   false, false);
    }
    // ---- causal mask on diagonal tiles ----
    const int sc0 = s0 + l15, sc1 = sc0 + 16;
    if (s0 + 31 > qt0) {
#pragma unroll
      for (int i = 0; i < 8; ++i) {
        int qr = qt0 + roff + i;
        if (sc0 > qr) sa[i] = -3.0e38f;
        if (sc1 > qr) sb[i] = -3.0e38f;
      }
    }
    // ---- online softmax (rows live in 16-lane groups; xor 1,2,4,8 stays in-group) ----
#pragma unroll
    for (int i = 0; i < 8; ++i) {
      float tm = fmaxf(sa[i], sb[i]);
      tm = fmaxf(tm, __shfl_xor(tm, 1, 32));
      tm = fmaxf(tm, __shfl_xor(tm, 2, 32));
      tm = fmaxf(tm, __shfl_xor(tm, 4, 32));
      tm = fmaxf(tm, __shfl_xor(tm, 8, 32));
      float mnew = fmaxf(mrun[i], tm);
      alpha[i] = __expf(mrun[i] - mnew);
      mrun[i] = mnew;
      float pa = __expf(sa[i] - mnew);
      float pb = __expf(sb[i] - mnew);
      sa[i] = pa; sb[i] = pb;
      float rs = pa + pb;
      rs += __shfl_xor(rs, 1, 32);
      rs += __shfl_xor(rs, 2, 32);
      rs += __shfl_xor(rs, 4, 32);
      rs += __shfl_xor(rs, 8, 32);
      lrun[i] = lrun[i] * alpha[i] + rs;
    }
    // ---- P (C-layout) -> LDS -> A-layout fragment ----
#pragma unroll
    for (int i = 0; i < 8; ++i) {
      int r = roff + i;
      pt[r * 48 + l15]      = f2bf(sa[i]);
      pt[r * 48 + 16 + l15] = f2bf(sb[i]);
    }
    asm volatile("s_wait_dscnt 0x0" ::: "memory");
    Frag16 pf;
    {
      const u16* pp = pt + l15 * 48 + aoff;
      pf.q[0] = *(const u32x4*)(pp);
      pf.q[1] = *(const u32x4*)(pp + 16);
    }
    // ---- O = alpha*O + P @ V  (V^T gives K-contiguous B fragments) ----
#pragma unroll
    for (int dc = 0; dc < 8; ++dc) {
      const u16* vp = Vthead + (size_t)(dc * 16 + l15) * Tk + s0 + aoff;
      Frag16 vf;
      vf.q[0] = *(const u32x4*)(vp);
      vf.q[1] = *(const u32x4*)(vp + 16);
      v8f oc = o[dc];
#pragma unroll
      for (int i = 0; i < 8; ++i) oc[i] *= alpha[i];
      o[dc] = __builtin_amdgcn_wmma_f32_16x16x32_bf16(false, pf.v, false, vf.v, (short)0,
                                                      oc, false, false);
    }
  }
  // ---- epilogue: normalize and store bf16 ----
#pragma unroll
  for (int dc = 0; dc < 8; ++dc) {
#pragma unroll
    for (int i = 0; i < 8; ++i) {
      int row = qt0 + roff + i;
      float v = o[dc][i] / lrun[i];
      Yb[((size_t)b * Tk + row) * Ck + h * Dk + dc * 16 + l15] = f2bf(v);
    }
  }
}

extern "C" void kernel_launch(void* const* d_in, const int* in_sizes, int n_in,
                              void* d_out, int out_size, void* d_ws, size_t ws_size,
                              hipStream_t stream) {
  const float* x   = (const float*)d_in[0];
  const float* Wq  = (const float*)d_in[1];
  const float* Wkv = (const float*)d_in[2];
  const float* Wc  = (const float*)d_in[3];
  const float* bc  = (const float*)d_in[4];
  float* out = (float*)d_out;

  char* ws = (char*)d_ws;
  size_t off = 0;
  auto alloc = [&](size_t bytes) {
    char* p = ws + off;
    off += (bytes + 255) & ~(size_t)255;
    return p;
  };
  u16* xb   = (u16*)alloc((size_t)BT * Ck * 2);          // x in bf16 (reused as Yb)
  u16* Wqt  = (u16*)alloc((size_t)Ck * Ck * 2);          // Wq^T  [N,K]
  u16* Wkvt = (u16*)alloc((size_t)KVC * Ck * 2);         // Wkv^T [N,K]
  u16* Wct  = (u16*)alloc((size_t)Ck * Ck * 2);          // Wc^T  [N,K]
  u16* Qb   = (u16*)alloc((size_t)BT * Ck * 2);          // Q bf16 (pre-scaled)
  u16* KVb  = (u16*)alloc((size_t)BT * KVC * 2);         // [K | V] bf16
  u16* Vt   = (u16*)alloc((size_t)Bk * HKVk * Dk * Tk * 2);  // V^T bf16
  u16* Yb   = xb;  // xb is dead after the two projection GEMMs
  if (off > ws_size) return;

  {
    int n = BT * Ck;
    cvt_f32_bf16<<<(n + 255) / 256, 256, 0, stream>>>(x, xb, n);
  }
  transpose_cvt<<<(Ck * Ck  + 255) / 256, 256, 0, stream>>>(Wq,  Wqt,  Ck, Ck);
  transpose_cvt<<<(Ck * KVC + 255) / 256, 256, 0, stream>>>(Wkv, Wkvt, Ck, KVC);
  transpose_cvt<<<(Ck * Ck  + 255) / 256, 256, 0, stream>>>(Wc,  Wct,  Ck, Ck);

  const float qscale = 0.08838834764831845f;  // 1/sqrt(128)
  gemm_wmma_kernel<0><<<dim3(Ck / 64, BT / 128), 256, 0, stream>>>(
      xb, Wqt, Qb, nullptr, nullptr, BT, Ck, Ck, qscale);
  gemm_wmma_kernel<0><<<dim3(KVC / 64, BT / 128), 256, 0, stream>>>(
      xb, Wkvt, KVb, nullptr, nullptr, BT, KVC, Ck, 1.0f);
  {
    int n = Bk * HKVk * Dk * Tk;
    transpose_v_kernel<<<(n + 255) / 256, 256, 0, stream>>>(KVb, Vt, n);
  }
  attn_kernel<<<dim3(Tk / 128, Hk, Bk), 256, 0, stream>>>(Qb, KVb, Vt, Yb);
  gemm_wmma_kernel<1><<<dim3(Ck / 64, BT / 128), 256, 0, stream>>>(
      Yb, Wct, nullptr, out, bc, BT, Ck, Ck, 1.0f);
}
